// MultiHeadAttentionLayer_83691732730598
// MI455X (gfx1250) — matrix operand order repaired and checked by reading, not compiled, not run on hardware
//
#include <hip/hip_runtime.h>
#include <hip/hip_bf16.h>

typedef _Float16 half_t;
typedef __attribute__((ext_vector_type(16))) _Float16 v16h;
typedef __attribute__((ext_vector_type(8)))  _Float16 v8h;
typedef __attribute__((ext_vector_type(4)))  _Float16 v4h;
typedef __attribute__((ext_vector_type(8)))  float    v8f;

#define DK_   64
#define NH_   16
#define DM_   1024
#define TS_   2048
#define NB_   2
#define MR_   (NB_*TS_)    /* 4096 rows total */

#define LOG2E 1.44269504088896340736f

// ---------------------------------------------------------------------------
// WMMA helper: D = A(16x32 f16) * B(32x16 f16) + C(16x16 f32)
// ---------------------------------------------------------------------------
__device__ __forceinline__ v8f wmma_f16(v16h a, v16h b, v8f c) {
    return __builtin_amdgcn_wmma_f32_16x16x32_f16(
        /*neg_a=*/false, a, /*neg_b=*/false, b,
        /*c_mod=*/(short)0, c, /*reuse_a=*/false, /*reuse_b=*/false);
}

// A-fragment (16x32, 16-bit) from row-major f16: lane l holds row M=l%16,
// K = khalf*16 + 0..7 (VGPR0-3) and K = khalf*16 + 8..15 (VGPR4-7),
// khalf = l/16.  Two contiguous 16B loads.
__device__ __forceinline__ v16h load_a_frag(const half_t* row, int kbase, int khalf) {
    v8h lo = *(const v8h*)(row + kbase + khalf * 16);
    v8h hi = *(const v8h*)(row + kbase + khalf * 16 + 8);
    return __builtin_shufflevector(lo, hi, 0,1,2,3,4,5,6,7,8,9,10,11,12,13,14,15);
}

__device__ __forceinline__ unsigned short h_bits(half_t h) {
    union { half_t h; unsigned short u; } c; c.h = h; return c.u;
}

// ---------------------------------------------------------------------------
// Pack W [K x N] f32 row-major -> B-fragment-ordered f16 dwords.
// ---------------------------------------------------------------------------
__global__ void pack_w(const float* __restrict__ W, unsigned int* __restrict__ dst,
                       int K, int N) {
    int d = blockIdx.x * 256 + threadIdx.x;
    int total = (K * N) >> 1;
    if (d >= total) return;
    int j    = d & 7;
    int lane = (d >> 3) & 31;
    int tile = d >> 8;
    int ntiles = N >> 4;
    int nt = tile % ntiles, kt = tile / ntiles;
    int n  = nt * 16 + (lane & 15);
    int k0 = kt * 32 + ((lane >> 4) & 1) * 16 + 2 * j;
    unsigned int lo = h_bits((half_t)W[(size_t)k0 * N + n]);
    unsigned int hi = h_bits((half_t)W[(size_t)(k0 + 1) * N + n]);
    dst[d] = lo | (hi << 16);
}

// f32 -> f16 row-major conversion (float4 vectorized)
__global__ void cvt_f16(const float4* __restrict__ s, half_t* __restrict__ d, int n4) {
    int i = blockIdx.x * 256 + threadIdx.x;
    if (i < n4) {
        float4 f = s[i];
        v4h h = { (half_t)f.x, (half_t)f.y, (half_t)f.z, (half_t)f.w };
        *(v4h*)(d + (size_t)i * 4) = h;
    }
}

// Pack key-padding mask into one 32-bit word per 32-key tile.
__global__ void pack_mask(const int* __restrict__ m, unsigned int* __restrict__ bits,
                          int nwords) {
    int i = blockIdx.x * 256 + threadIdx.x;
    if (i < nwords) {
        unsigned int w = 0;
        for (int b2 = 0; b2 < 32; ++b2) w |= (m[i * 32 + b2] != 0 ? 1u : 0u) << b2;
        bits[i] = w;
    }
}

// ---------------------------------------------------------------------------
// GEMM stage helpers (32x64 wave tile, ping-pong double buffered)
// ---------------------------------------------------------------------------
__device__ __forceinline__ void load_stage(const half_t* xrow0, const half_t* xrow1,
        const unsigned int* wp, int kt, int khalf, v16h& a0, v16h& a1, v16h b[4]) {
    a0 = load_a_frag(xrow0, kt * 32, khalf);
    a1 = load_a_frag(xrow1, kt * 32, khalf);
#pragma unroll
    for (int ns = 0; ns < 4; ++ns)
        b[ns] = *(const v16h*)(wp + ((size_t)kt * 64 + ns) * 256);
}

__device__ __forceinline__ void mma_stage(v8f acc[8], const v16h& a0, const v16h& a1,
                                          const v16h b[4]) {
#pragma unroll
    for (int ns = 0; ns < 4; ++ns) {
        acc[ns]     = wmma_f16(a0, b[ns], acc[ns]);
        acc[4 + ns] = wmma_f16(a1, b[ns], acc[4 + ns]);
    }
}

__device__ __forceinline__ void mma_stage_zero(v8f acc[8], const v16h& a0, const v16h& a1,
                                               const v16h b[4]) {
    const v8f z = {};
#pragma unroll
    for (int ns = 0; ns < 4; ++ns) {
        acc[ns]     = wmma_f16(a0, b[ns], z);   // SRC2 folds to inline 0
        acc[4 + ns] = wmma_f16(a1, b[ns], z);
    }
}

// ---------------------------------------------------------------------------
// Projection GEMM: out = (X[4096x1024]f16 @ Wpk + bias) * scale.
// MODE 0: row-major f16 per head [b][h][t][dk]        (used for K)
// MODE 1: B-frag packed per head over (K=dk, N=T)      (used for Q, scale=log2e/8)
// MODE 2: B-frag packed per head over (K=T, N=dk)      (used for V)
// ---------------------------------------------------------------------------
template<int MODE>
__global__ void __launch_bounds__(256) gemm_proj(
    const half_t* __restrict__ X, const unsigned int* __restrict__ Wpk,
    const float* __restrict__ bias, half_t* __restrict__ out, float scale)
{
    const int lane  = threadIdx.x & 31;
    const int lc    = lane & 15;
    const int khalf = lane >> 4;
    const int wid   = blockIdx.x * 8 + (threadIdx.x >> 5);
    const int mt    = wid >> 4;    // 0..127 (32 rows each)
    const int nt4   = wid & 15;    // 0..15 -> 64 output columns

    const half_t* xrow0 = X + (size_t)(mt * 32 + lc) * DM_;
    const half_t* xrow1 = xrow0 + (size_t)16 * DM_;
    const unsigned int* wp = Wpk + (size_t)nt4 * 4 * 256 + lane * 8;

    v8f acc[8];
    v16h a0A, a1A, bA[4];
    v16h a0B, a1B, bB[4];
    load_stage(xrow0, xrow1, wp, 0, khalf, a0A, a1A, bA);
    load_stage(xrow0, xrow1, wp, 1, khalf, a0B, a1B, bB);
    mma_stage_zero(acc, a0A, a1A, bA);                 // kt = 0
    for (int kt = 1; kt + 2 < DM_ / 32; kt += 2) {     // kt = 1,3,...,29
        load_stage(xrow0, xrow1, wp, kt + 1, khalf, a0A, a1A, bA);
        mma_stage(acc, a0B, a1B, bB);                  // kt
        load_stage(xrow0, xrow1, wp, kt + 2, khalf, a0B, a1B, bB);
        mma_stage(acc, a0A, a1A, bA);                  // kt + 1
    }
    mma_stage(acc, a0B, a1B, bB);                      // kt = 31

#pragma unroll
    for (int mh = 0; mh < 2; ++mh) {
#pragma unroll
        for (int ns = 0; ns < 4; ++ns) {
            const int ng = nt4 * 64 + ns * 16 + lc;
            const float bn = bias[ng];
            const int hh = ng >> 6, dk = ng & 63;
#pragma unroll
            for (int r = 0; r < 8; ++r) {
                const int mg = mt * 32 + mh * 16 + r + 8 * khalf;
                const int bb = mg >> 11, t = mg & (TS_ - 1);
                const float val = (acc[mh * 4 + ns][r] + bn) * scale;
                if (MODE == 0) {
                    out[(((size_t)(bb * NH_ + hh) * TS_) + t) * DK_ + dk] = (half_t)val;
                } else if (MODE == 1) {
                    const size_t basebh = (size_t)(bb * NH_ + hh) * (TS_ * DK_);
                    const int kt2 = dk >> 5, j = (dk & 15) >> 1, th = dk & 1;
                    const int dl = ((dk >> 4) & 1) * 16 + (t & 15);
                    const int tt = t >> 4;
                    out[basebh + ((((size_t)kt2 * (TS_ / 16) + tt) * 32 + dl) * 8 + j) * 2 + th] = (half_t)val;
                } else {
                    const size_t basebh = (size_t)(bb * NH_ + hh) * (TS_ * DK_);
                    const int ktv = t >> 5, j = ((t & 7) >> 1) + 4 * ((t >> 4) & 1), th = t & 1;
                    const int dl = ((t >> 3) & 1) * 16 + (dk & 15);
                    const int ntile = dk >> 4;
                    out[basebh + ((((size_t)ktv * 4 + ntile) * 32 + dl) * 8 + j) * 2 + th] = (half_t)val;
                }
            }
        }
    }
}

// Output projection: out_f32 = A[4096x1024]f16 @ Wo_pk + bo
__global__ void __launch_bounds__(256) gemm_out(
    const half_t* __restrict__ X, const unsigned int* __restrict__ Wpk,
    const float* __restrict__ bias, float* __restrict__ out)
{
    const int lane  = threadIdx.x & 31;
    const int lc    = lane & 15;
    const int khalf = lane >> 4;
    const int wid   = blockIdx.x * 8 + (threadIdx.x >> 5);
    const int mt    = wid >> 4;
    const int nt4   = wid & 15;

    const half_t* xrow0 = X + (size_t)(mt * 32 + lc) * DM_;
    const half_t* xrow1 = xrow0 + (size_t)16 * DM_;
    const unsigned int* wp = Wpk + (size_t)nt4 * 4 * 256 + lane * 8;

    v8f acc[8];
    v16h a0A, a1A, bA[4];
    v16h a0B, a1B, bB[4];
    load_stage(xrow0, xrow1, wp, 0, khalf, a0A, a1A, bA);
    load_stage(xrow0, xrow1, wp, 1, khalf, a0B, a1B, bB);
    mma_stage_zero(acc, a0A, a1A, bA);
    for (int kt = 1; kt + 2 < DM_ / 32; kt += 2) {
        load_stage(xrow0, xrow1, wp, kt + 1, khalf, a0A, a1A, bA);
        mma_stage(acc, a0B, a1B, bB);
        load_stage(xrow0, xrow1, wp, kt + 2, khalf, a0B, a1B, bB);
        mma_stage(acc, a0A, a1A, bA);
    }
    mma_stage(acc, a0B, a1B, bB);

#pragma unroll
    for (int mh = 0; mh < 2; ++mh) {
#pragma unroll
        for (int ns = 0; ns < 4; ++ns) {
            const int ng = nt4 * 64 + ns * 16 + lc;
            const float bn = bias[ng];
#pragma unroll
            for (int r = 0; r < 8; ++r) {
                const int mg = mt * 32 + mh * 16 + r + 8 * khalf;
                out[(size_t)mg * DM_ + ng] = acc[mh * 4 + ns][r] + bn;
            }
        }
    }
}

// ---------------------------------------------------------------------------
// Flash attention (transposed-scores formulation, log2-domain softmax).
// S^T = K_tile @ Q^T: C-frag has lane = query, VGPR = key, so the softmax row
// reduction is lane-local (+1 shfl_xor(16)) and exp2(S^T) is ALREADY the
// A-fragment of P for the P@V WMMA -- no LDS transpose needed at all.
// log2e is folded into the Q projection scale; __builtin_amdgcn_exp2f maps
// 1:1 to v_exp_f32 (no libm range-fixup code, no pre-multiply).
// ---------------------------------------------------------------------------
__device__ __forceinline__ void load_kafrags(const half_t* krow, int ktile,
        int lc, int khalf, v16h aK[4]) {
    const half_t* r0 = krow + (size_t)(ktile * 32 + lc) * DK_;
    const half_t* r1 = krow + (size_t)(ktile * 32 + 16 + lc) * DK_;
    aK[0] = load_a_frag(r0, 0,  khalf);
    aK[1] = load_a_frag(r0, 32, khalf);
    aK[2] = load_a_frag(r1, 0,  khalf);
    aK[3] = load_a_frag(r1, 32, khalf);
}

__device__ __forceinline__ void attn_step(int ktile, const v16h& bq0, const v16h& bq1,
        const v16h aK[4], const unsigned int* vbp, const unsigned int* mb,
        int khalf, v8f o[4], float& mrun, float& lrun)
{
    // prefetch V fragments for this key-tile (consumed after softmax)
    v16h bv[4];
#pragma unroll
    for (int ns = 0; ns < 4; ++ns)
        bv[ns] = *(const v16h*)(vbp + ((size_t)ktile * 4 + ns) * 256);

    const v8f z = {};
    v8f st0 = wmma_f16(aK[0], bq0, z);   // keys base+0..15 (split by lane half)
    st0 = wmma_f16(aK[1], bq1, st0);
    v8f st1 = wmma_f16(aK[2], bq0, z);   // keys base+16..31
    st1 = wmma_f16(aK[3], bq1, st1);

    // mask: bit k of w = key (ktile*32 + k) valid   (scores are in log2 domain)
    const unsigned int w = mb[ktile];
    float sv[16];
#pragma unroll
    for (int r = 0; r < 8; ++r) {
        sv[r]     = st0[r] + (((w >> (khalf * 8 + r)) & 1u)      ? 0.0f : -1e30f);
        sv[8 + r] = st1[r] + (((w >> (16 + khalf * 8 + r)) & 1u) ? 0.0f : -1e30f);
    }
    // lane-local max over 16 keys, then combine lane halves
    float tmax = sv[0];
#pragma unroll
    for (int e = 1; e < 16; ++e) tmax = fmaxf(tmax, sv[e]);
    tmax = fmaxf(tmax, __shfl_xor(tmax, 16, 32));
    const float mnew = fmaxf(mrun, tmax);
    const float corr = __builtin_amdgcn_exp2f(mrun - mnew);   // v_exp_f32
    // exp2 -> P fragment (register-resident A-frag) + lane-local sum
    v16h pa;
    float psum = 0.0f;
#pragma unroll
    for (int e = 0; e < 16; ++e) {
        const float p = __builtin_amdgcn_exp2f(sv[e] - mnew); // v_exp_f32
        psum += p;
        pa[e] = (half_t)p;
    }
    psum += __shfl_xor(psum, 16, 32);
    lrun = lrun * corr + psum;
    mrun = mnew;
    // broadcast per-query corr (held at lane = query) to O's per-VGPR queries
#pragma unroll
    for (int r = 0; r < 8; ++r) {
        const float cr = __shfl(corr, khalf * 8 + r, 32);
#pragma unroll
        for (int ns = 0; ns < 4; ++ns) o[ns][r] *= cr;
    }
#pragma unroll
    for (int ns = 0; ns < 4; ++ns)
        o[ns] = wmma_f16(pa, bv[ns], o[ns]);
}

__global__ void __launch_bounds__(256) attn_kernel(
    const half_t* __restrict__ Kr, const unsigned int* __restrict__ Qp,
    const unsigned int* __restrict__ Vp, const unsigned int* __restrict__ mbits,
    half_t* __restrict__ A)
{
    const int lane  = threadIdx.x & 31;
    const int lc    = lane & 15;
    const int khalf = lane >> 4;
    const int wid   = blockIdx.x * 8 + (threadIdx.x >> 5);
    const int qt = wid & 127;
    const int hh = (wid >> 7) & 15;
    const int bb = wid >> 11;

    // Q^T B-fragments for this query tile (loaded once; scale folded in proj)
    const unsigned int* qb = Qp + (size_t)(bb * NH_ + hh) * (TS_ * DK_ / 2) + lane * 8;
    const v16h bq0 = *(const v16h*)(qb + (size_t)qt * 256);           // dk 0..31
    const v16h bq1 = *(const v16h*)(qb + (size_t)(128 + qt) * 256);   // dk 32..63
    const half_t* krow = Kr + (size_t)(bb * NH_ + hh) * TS_ * DK_;
    const unsigned int* vbp = Vp + (size_t)(bb * NH_ + hh) * (TS_ * DK_ / 2) + lane * 8;
    const unsigned int* mb = mbits + bb * (TS_ / 32);

    v8f o[4] = {{}, {}, {}, {}};
    float mrun = -1e30f, lrun = 0.0f;

    v16h aKA[4], aKB[4];
    load_kafrags(krow, 0, lc, khalf, aKA);
    for (int t = 0; t + 2 < TS_ / 32; t += 2) {        // t = 0,2,...,60
        load_kafrags(krow, t + 1, lc, khalf, aKB);
        attn_step(t,     bq0, bq1, aKA, vbp, mb, khalf, o, mrun, lrun);
        load_kafrags(krow, t + 2, lc, khalf, aKA);
        attn_step(t + 1, bq0, bq1, aKB, vbp, mb, khalf, o, mrun, lrun);
    }
    load_kafrags(krow, TS_ / 32 - 1, lc, khalf, aKB);
    attn_step(TS_ / 32 - 2, bq0, bq1, aKA, vbp, mb, khalf, o, mrun, lrun);
    attn_step(TS_ / 32 - 1, bq0, bq1, aKB, vbp, mb, khalf, o, mrun, lrun);

    // normalize: 1/l per query broadcast to per-VGPR rows, store A row-major
    const float linv = 1.0f / lrun;
#pragma unroll
    for (int r = 0; r < 8; ++r) {
        const float ir = __shfl(linv, khalf * 8 + r, 32);
        const int mg = bb * TS_ + qt * 16 + r + 8 * khalf;
#pragma unroll
        for (int ns = 0; ns < 4; ++ns)
            A[(size_t)mg * DM_ + hh * 64 + ns * 16 + lc] = (half_t)(o[ns][r] * ir);
    }
}

// ---------------------------------------------------------------------------
extern "C" void kernel_launch(void* const* d_in, const int* in_sizes, int n_in,
                              void* d_out, int out_size, void* d_ws, size_t ws_size,
                              hipStream_t stream) {
    const float* q  = (const float*)d_in[0];
    const float* k  = (const float*)d_in[1];
    const float* v  = (const float*)d_in[2];
    const int*   mk = (const int*)d_in[3];
    const float* Wq = (const float*)d_in[4];
    const float* bq = (const float*)d_in[5];
    const float* Wk = (const float*)d_in[6];
    const float* bk = (const float*)d_in[7];
    const float* Wv = (const float*)d_in[8];
    const float* bv = (const float*)d_in[9];
    const float* Wo = (const float*)d_in[10];
    const float* bo = (const float*)d_in[11];
    float* out = (float*)d_out;

    char* ws = (char*)d_ws;
    const size_t SZ_XH  = (size_t)MR_ * DM_ * 2;   // 8 MB f16 activation
    const size_t SZ_WPK = (size_t)DM_ * DM_ * 2;   // 2 MB packed weights
    half_t*       Xq   = (half_t*)(ws + 0 * SZ_XH);
    half_t*       Xk   = (half_t*)(ws + 1 * SZ_XH);
    half_t*       Xv   = (half_t*)(ws + 2 * SZ_XH);
    unsigned int* Wqpk = (unsigned int*)(ws + 3 * SZ_XH);
    unsigned int* Wkpk = (unsigned int*)(ws + 3 * SZ_XH + 1 * SZ_WPK);
    unsigned int* Wvpk = (unsigned int*)(ws + 3 * SZ_XH + 2 * SZ_WPK);
    unsigned int* Wopk = (unsigned int*)(ws + 3 * SZ_XH + 3 * SZ_WPK);
    char* p2 = ws + 3 * SZ_XH + 4 * SZ_WPK;
    half_t* Qpk = (half_t*)(p2 + 0 * SZ_XH);   // packed Q (B-frag dk x T)
    half_t* Krm = (half_t*)(p2 + 1 * SZ_XH);   // row-major K per head
    half_t* Vpk = (half_t*)(p2 + 2 * SZ_XH);   // packed V (B-frag T x dk)
    half_t* Ab  = (half_t*)(p2 + 3 * SZ_XH);
    unsigned int* Mb = (unsigned int*)(p2 + 4 * SZ_XH);  // 512 B mask bits

    dim3 blk(256);
    // pack weights (DM*DM/2 dwords each -> 2048 blocks)
    pack_w<<<2048, blk, 0, stream>>>(Wq, Wqpk, DM_, DM_);
    pack_w<<<2048, blk, 0, stream>>>(Wk, Wkpk, DM_, DM_);
    pack_w<<<2048, blk, 0, stream>>>(Wv, Wvpk, DM_, DM_);
    pack_w<<<2048, blk, 0, stream>>>(Wo, Wopk, DM_, DM_);
    // f32 -> f16 activations (4M elems / 4 per thread -> 4096 blocks)
    cvt_f16<<<4096, blk, 0, stream>>>((const float4*)q, Xq, MR_ * DM_ / 4);
    cvt_f16<<<4096, blk, 0, stream>>>((const float4*)k, Xk, MR_ * DM_ / 4);
    cvt_f16<<<4096, blk, 0, stream>>>((const float4*)v, Xv, MR_ * DM_ / 4);
    // mask bits: 2*2048/32 = 128 words
    pack_mask<<<1, blk, 0, stream>>>(mk, Mb, NB_ * TS_ / 32);
    // projections: 128 mt * 16 nt4 = 2048 waves / 8 = 256 blocks
    // Q: packed layout, scale = log2e / sqrt(dk)  (log2-domain softmax)
    gemm_proj<1><<<256, blk, 0, stream>>>(Xq, Wqpk, bq, Qpk, 0.125f * LOG2E);
    gemm_proj<0><<<256, blk, 0, stream>>>(Xk, Wkpk, bk, Krm, 1.0f);    // K row-major
    gemm_proj<2><<<256, blk, 0, stream>>>(Xv, Wvpk, bv, Vpk, 1.0f);    // V packed
    // attention: 2*16*128 = 4096 waves / 8 = 512 blocks
    attn_kernel<<<512, blk, 0, stream>>>(Krm, (const unsigned int*)Qpk,
                                         (const unsigned int*)Vpk, Mb, Ab);
    // output projection -> f32 d_out
    gemm_out<<<256, blk, 0, stream>>>(Ab, Wopk, bo, out);
}
